// EncoderDecoderConvLSTM_20916490732162
// MI455X (gfx1250) — matrix operand order repaired and checked by reading, not compiled, not run on hardware
//
#include <hip/hip_runtime.h>
#include <hip/hip_bf16.h>

typedef __attribute__((ext_vector_type(16))) __bf16 v16bf;
typedef __attribute__((ext_vector_type(8)))  __bf16 bhalf8;
typedef __attribute__((ext_vector_type(4)))  __bf16 bhalf4;
typedef __attribute__((ext_vector_type(8)))  float  v8f;

#define H_DIM 128
#define W_DIM 128
#define B_DIM 8
#define T_DIM 12
#define PLANE (H_DIM * W_DIM)

// ---------------------------------------------------------------------------
// Weight prep: fp32 (Cout, Cin, 3, 3) -> bf16, packed in a WMMA
// B-matrix (32x16 K-major) register layout (contiguous k per lane half, same as A) with TAP-MAJOR k ordering:
//   k = tap*CinP + cin   (tap = dy*3+dx, CinP = Cin padded to mult. of 8)
//   idx = ((nt*KB + kb)*32 + lane)*16 + j
//   column n = nt*16 + (lane&15),  k = kb*32 + (lane>>4)*16 + j
// ---------------------------------------------------------------------------
__global__ void prep_weights(const float* __restrict__ Wsrc,
                             __bf16* __restrict__ Bprep,
                             int Cout, int Cin, int CinP, int NT, int KB) {
    int total = NT * KB * 512;
    int idx = blockIdx.x * blockDim.x + threadIdx.x;
    if (idx >= total) return;
    int j    = idx & 15;
    int lane = (idx >> 4) & 31;
    int kb   = (idx >> 9) % KB;
    int nt   = idx / (512 * KB);
    int n = nt * 16 + (lane & 15);
    int k = kb * 32 + (lane >> 4) * 16 + j;
    int tap = k / CinP;
    int cin = k % CinP;
    float v = 0.f;
    if (n < Cout && tap < 9 && cin < Cin)
        v = Wsrc[(n * Cin + cin) * 9 + tap];
    Bprep[idx] = (__bf16)v;
}

// ---------------------------------------------------------------------------
// Fused ConvLSTM cell: z = conv3x3(concat[A, H], W)+b, then LSTM gate math.
// One block = 128 threads = 4 waves; handles (batch b, row y, 64-px segment).
// Wave w = (m-half, n-half): pixels (w>>1)*32..+31 (TWO 16-px A tiles) and
// N-tiles (w&1)*NT/2.. — each B fragment feeds two WMMAs (halved B traffic,
// 2x independent accumulators to hide WMMA RAW hazards).
// Staged input: sA[ry 0..3][col 0..67][cinP]  (bf16, row 3 = zeros for K-pad)
// A fragment = two contiguous 16B LDS chunks (8-run of k within one tap).
// ---------------------------------------------------------------------------
template<int ChA, int Ch>
__global__ __launch_bounds__(128)
void convlstm_cell(const float* __restrict__ inA, long strideAb,
                   const float* __restrict__ inH,
                   const __bf16* __restrict__ Bprep,
                   const float* __restrict__ bias,
                   float* __restrict__ Cst,
                   float* __restrict__ Hout) {
    constexpr int ChB  = Ch;
    constexpr int Cin  = ChA + ChB;
    constexpr int CinP = (Cin + 7) & ~7;         // 16 / 24 / 48 / 48
    constexpr int Cout = 4 * Ch;
    constexpr int KB   = (9 * CinP + 31) / 32;
    constexpr int NT   = Cout / 16;
    constexpr int NT2  = NT / 2;                 // n-tiles per wave
    constexpr int SCOL = 68;                     // staged cols per row
    constexpr int ZP   = Cout + 4;               // padded z stride (floats)
    constexpr int STAGE_BYTES = 4 * SCOL * CinP * 2;
    constexpr int Z_BYTES     = 64 * ZP * 4;
    constexpr int SMEM_BYTES  = STAGE_BYTES > Z_BYTES ? STAGE_BYTES : Z_BYTES;

    __shared__ alignas(16) unsigned char smem[SMEM_BYTES];
    __bf16* sA = (__bf16*)smem;
    float*  zs = (float*)smem;

    const int bx   = blockIdx.x;
    const int xseg = bx & 1;
    const int y    = (bx >> 1) & (H_DIM - 1);
    const int b    = bx >> 8;                    // / (2*H_DIM)
    const int x0   = xseg * 64;
    const int tid  = threadIdx.x;

    // ---- stage concat[A,H]: rows y-1..y+1 (+ zero row 3), cols x0-1..x0+64,
    //      channel-minor bf16, 4 channels packed per ds_store_b64 -----------
    constexpr int CG = CinP / 4;
    const int NS = 4 * 66 * CG;
    for (int i = tid; i < NS; i += 128) {
        int col = i % 66;
        int cg  = (i / 66) % CG;
        int ry  = i / (66 * CG);
        int gy  = y + ry - 1;
        int gx  = x0 + col - 1;
        bhalf4 pk = {};
        if (ry < 3 && gy >= 0 && gy < H_DIM && gx >= 0 && gx < W_DIM) {
#pragma unroll
            for (int l = 0; l < 4; ++l) {
                int cin = cg * 4 + l;
                float v = 0.f;
                if (cin < Cin) {
                    if (cin < ChA)
                        v = inA[(long)b * strideAb + (long)cin * PLANE + gy * W_DIM + gx];
                    else
                        v = inH[(((long)b * ChB + (cin - ChA)) * H_DIM + gy) * W_DIM + gx];
                }
                pk[l] = (__bf16)v;
            }
        }
        *(bhalf4*)(sA + ((ry * SCOL + col) * CinP + cg * 4)) = pk;
    }
    __syncthreads();

    const int lane  = tid & 31;
    const int wave  = tid >> 5;
    const int wavem = wave >> 1;                 // m-half: pixels wavem*32..
    const int ntb   = (wave & 1) * NT2;          // first n-tile of this wave
    const int half  = lane >> 4;
    const int mrow  = lane & 15;
    const int px    = wavem * 32 + mrow;         // first A-tile pixel row
    const int pxoff = px * (CinP * 2);           // byte offset within stage row
    constexpr int MT_OFF = 16 * CinP * 2;        // second A-tile: +16 pixels

    v8f acc0[NT2], acc1[NT2];
#pragma unroll
    for (int nt = 0; nt < NT2; ++nt) {
        acc0[nt] = (v8f){0.f,0.f,0.f,0.f,0.f,0.f,0.f,0.f};
        acc1[nt] = (v8f){0.f,0.f,0.f,0.f,0.f,0.f,0.f,0.f};
    }

    const v16bf* Bv = (const v16bf*)Bprep;
    const char* sbase = (const char*)smem;

#pragma unroll
    for (int kb = 0; kb < KB; ++kb) {
        // byte offset of the 16B chunk holding k0..k0+7 (within one tap)
        auto chunk_off = [](int k0) {
            int tap = k0 / CinP, cin = k0 % CinP;
            int dy = tap / 3, dx = tap % 3;
            return ((dy * SCOL + dx) * CinP + cin) * 2;
        };
        const int ol0 = chunk_off(kb * 32 + 0);       // lo, half=0
        const int ol1 = chunk_off(kb * 32 + 16);      // lo, half=1
        const int oh0 = chunk_off(kb * 32 + 8);       // hi, half=0
        const int oh1 = chunk_off(kb * 32 + 24);      // hi, half=1
        const int sl  = (half ? ol1 : ol0) + pxoff;
        const int sh  = (half ? oh1 : oh0) + pxoff;
        bhalf8 lo0 = *(const bhalf8*)(sbase + sl);
        bhalf8 hi0 = *(const bhalf8*)(sbase + sh);
        bhalf8 lo1 = *(const bhalf8*)(sbase + sl + MT_OFF);
        bhalf8 hi1 = *(const bhalf8*)(sbase + sh + MT_OFF);
        v16bf a0 = __builtin_shufflevector(lo0, hi0,
                     0,1,2,3,4,5,6,7,8,9,10,11,12,13,14,15);
        v16bf a1 = __builtin_shufflevector(lo1, hi1,
                     0,1,2,3,4,5,6,7,8,9,10,11,12,13,14,15);
#pragma unroll
        for (int nt = 0; nt < NT2; ++nt) {
            v16bf bf = Bv[((ntb + nt) * KB + kb) * 32 + lane];
            acc0[nt] = __builtin_amdgcn_wmma_f32_16x16x32_bf16(
                false, a0, false, bf, (short)0, acc0[nt], false, false);
            acc1[nt] = __builtin_amdgcn_wmma_f32_16x16x32_bf16(
                false, a1, false, bf, (short)0, acc1[nt], false, false);
        }
    }
    __syncthreads();   // stage buffer no longer needed; reuse as z tile

    // ---- write z to LDS (C/D layout: lane holds column n, 8 rows) ---------
#pragma unroll
    for (int nt = 0; nt < NT2; ++nt) {
        int n = (ntb + nt) * 16 + mrow;
#pragma unroll
        for (int r = 0; r < 8; ++r) {
            int m = r + 8 * half;
            zs[(wavem * 32 + m) * ZP + n]      = acc0[nt][r];
            zs[(wavem * 32 + 16 + m) * ZP + n] = acc1[nt][r];
        }
    }
    __syncthreads();

    // ---- LSTM gate fusion (fp32 states, c in-place, h ping-pong) ----------
    for (int idx = tid; idx < 64 * Ch; idx += 128) {
        int ch  = idx % Ch;
        int pix = idx / Ch;
        int x   = x0 + pix;
        const float* zrow = zs + pix * ZP;
        float zi = zrow[ch]          + bias[ch];
        float zf = zrow[Ch + ch]     + bias[Ch + ch];
        float zo = zrow[2 * Ch + ch] + bias[2 * Ch + ch];
        float zg = zrow[3 * Ch + ch] + bias[3 * Ch + ch];
        float ig = 1.f / (1.f + __expf(-zi));
        float fg = 1.f / (1.f + __expf(-zf));
        float og = 1.f / (1.f + __expf(-zo));
        float g  = tanhf(zg);
        long cidx = (((long)b * Ch + ch) * H_DIM + y) * W_DIM + x;
        float c2 = fg * Cst[cidx] + ig * g;
        Cst[cidx]  = c2;
        Hout[cidx] = og * tanhf(c2);
    }
}

// ---------------------------------------------------------------------------
// Final 16->1 conv + sigmoid, written straight into the output slice for t.
// ---------------------------------------------------------------------------
__global__ void final_conv(const float* __restrict__ h3,
                           const float* __restrict__ Wd,
                           const float* __restrict__ bd,
                           float* __restrict__ out, int t) {
    int idx = blockIdx.x * blockDim.x + threadIdx.x;
    if (idx >= B_DIM * PLANE) return;
    int x = idx & (W_DIM - 1);
    int y = (idx >> 7) & (H_DIM - 1);
    int b = idx >> 14;
    float s = bd[0];
    for (int ch = 0; ch < 16; ++ch) {
        const float* base = h3 + ((long)b * 16 + ch) * PLANE;
#pragma unroll
        for (int dy = 0; dy < 3; ++dy) {
            int gy = y + dy - 1;
            if (gy < 0 || gy >= H_DIM) continue;
#pragma unroll
            for (int dx = 0; dx < 3; ++dx) {
                int gx = x + dx - 1;
                if (gx < 0 || gx >= W_DIM) continue;
                s += base[gy * W_DIM + gx] * Wd[ch * 9 + dy * 3 + dx];
            }
        }
    }
    out[((long)b * T_DIM + t) * PLANE + y * W_DIM + x] = 1.f / (1.f + __expf(-s));
}

// ---------------------------------------------------------------------------
extern "C" void kernel_launch(void* const* d_in, const int* in_sizes, int n_in,
                              void* d_out, int out_size, void* d_ws, size_t ws_size,
                              hipStream_t stream) {
    (void)in_sizes; (void)n_in; (void)out_size; (void)ws_size;
    const float* x  = (const float*)d_in[0];
    const float* W0 = (const float*)d_in[1];
    const float* b0 = (const float*)d_in[2];
    const float* W1 = (const float*)d_in[3];
    const float* b1 = (const float*)d_in[4];
    const float* W2 = (const float*)d_in[5];
    const float* b2 = (const float*)d_in[6];
    const float* W3 = (const float*)d_in[7];
    const float* b3 = (const float*)d_in[8];
    const float* Wd = (const float*)d_in[9];
    const float* bd = (const float*)d_in[10];
    float* out = (float*)d_out;

    // ---- carve workspace: states (fp32, zeroed) then packed weights -------
    float* ws = (float*)d_ws;
    const long S8  = (long)B_DIM *  8 * PLANE;
    const long S16 = (long)B_DIM * 16 * PLANE;
    const long S32 = (long)B_DIM * 32 * PLANE;
    float* h0[2] = { ws, ws + S8 };  ws += 2 * S8;
    float* c0 = ws;                  ws += S8;
    float* h1[2] = { ws, ws + S16 }; ws += 2 * S16;
    float* c1 = ws;                  ws += S16;
    float* h2[2] = { ws, ws + S32 }; ws += 2 * S32;
    float* c2 = ws;                  ws += S32;
    float* h3[2] = { ws, ws + S16 }; ws += 2 * S16;
    float* c3 = ws;                  ws += S16;
    size_t stateBytes = (size_t)((char*)ws - (char*)d_ws);
    hipMemsetAsync(d_ws, 0, stateBytes, stream);

    // KB with padded channels: cell0 CinP=16 -> KB=5; cell1 24 -> 7; 48 -> 14
    __bf16* wp  = (__bf16*)ws;
    __bf16* B0p = wp; wp += 2 * 5  * 512;
    __bf16* B1p = wp; wp += 4 * 7  * 512;
    __bf16* B2p = wp; wp += 8 * 14 * 512;
    __bf16* B3p = wp; wp += 4 * 14 * 512;

    auto prep = [&](const float* W, __bf16* dst, int Cout, int Cin, int CinP,
                    int NT, int KB) {
        int total = NT * KB * 512;
        prep_weights<<<(total + 255) / 256, 256, 0, stream>>>(W, dst, Cout, Cin,
                                                              CinP, NT, KB);
    };
    prep(W0, B0p, 32,  9,  16, 2, 5);
    prep(W1, B1p, 64,  24, 24, 4, 7);
    prep(W2, B2p, 128, 48, 48, 8, 14);
    prep(W3, B3p, 64,  48, 48, 4, 14);

    dim3 grid(B_DIM * H_DIM * 2), blk(128);

    // ---- encoder ----------------------------------------------------------
    int p0 = 0, p1 = 0;
    for (int t = 0; t < T_DIM; ++t) {
        convlstm_cell<1, 8><<<grid, blk, 0, stream>>>(
            x + (long)t * PLANE, (long)T_DIM * PLANE,
            h0[p0], B0p, b0, c0, h0[p0 ^ 1]);
        p0 ^= 1;
        convlstm_cell<8, 16><<<grid, blk, 0, stream>>>(
            h0[p0], 8L * PLANE,
            h1[p1], B1p, b1, c1, h1[p1 ^ 1]);
        p1 ^= 1;
    }

    // ---- decoder + fused output conv --------------------------------------
    int p2 = 0, p3 = 0;
    for (int t = 0; t < T_DIM; ++t) {
        const float* ev = (t == 0) ? h1[p1] : h3[p3];
        convlstm_cell<16, 32><<<grid, blk, 0, stream>>>(
            ev, 16L * PLANE,
            h2[p2], B2p, b2, c2, h2[p2 ^ 1]);
        p2 ^= 1;
        convlstm_cell<32, 16><<<grid, blk, 0, stream>>>(
            h2[p2], 32L * PLANE,
            h3[p3], B3p, b3, c3, h3[p3 ^ 1]);
        p3 ^= 1;
        final_conv<<<(B_DIM * PLANE + 255) / 256, 256, 0, stream>>>(
            h3[p3], Wd, bd, out, t);
    }
}
